// GraphEncoder_23983097381433
// MI455X (gfx1250) — matrix operand order repaired and checked by reading, not compiled, not run on hardware
//
#include <hip/hip_runtime.h>

typedef __attribute__((ext_vector_type(16))) _Float16 v16h;
typedef __attribute__((ext_vector_type(8)))  float    v8f;

#define CHANNEL 512
#define EDGE_F  30
#define NB      16
#define NA      128

// Branchless SELU: exp computed unconditionally on min(x,0) (no overflow),
// result blended with v_cndmask instead of EXEC branches.
__device__ __forceinline__ float selu_f(float x) {
    const float l  = 1.0507009873554805f;
    const float la = 1.7580993408473766f;          // l * alpha
    float ex  = __expf(fminf(x, 0.0f));
    float neg = la * (ex - 1.0f);
    return x > 0.0f ? l * x : neg;
}

// ---------------- prep: f32 -> f16 weight conversion ----------------

// filter_W (512 x 30) -> padded f16 (512 x 32), zeros in k=30,31
__global__ __launch_bounds__(256) void cvt_filter_kernel(const float* __restrict__ fW,
                                                         _Float16* __restrict__ fWh) {
    int idx = blockIdx.x * blockDim.x + threadIdx.x;
    if (idx >= CHANNEL * 32) return;
    int c = idx >> 5, k = idx & 31;
    fWh[idx] = (k < EDGE_F) ? (_Float16)fW[c * EDGE_F + k] : (_Float16)0.0f;
}

// w_W (512 x 1024) -> f16 (512 x 1024)
__global__ __launch_bounds__(256) void cvt_w_kernel(const float* __restrict__ wW,
                                                    _Float16* __restrict__ wWh) {
    int idx = blockIdx.x * blockDim.x + threadIdx.x;
    if (idx < CHANNEL * 2 * CHANNEL) wWh[idx] = (_Float16)wW[idx];
}

// ---------------- kernel 1: fused edge filter + aggregation ----------------
// one wave per (b, i, c-tile).  h1[b,i,c] = sum_j selu(e[b,i,j,:]@Wf.T + bf)[c]
//                                          * edge_mask[b,i,j] * h[b,j,c]
__global__ __launch_bounds__(256) void edge_agg_kernel(const float* __restrict__ e,
                                                       const float* __restrict__ h,
                                                       const float* __restrict__ edge_mask,
                                                       const float* __restrict__ filter_b,
                                                       const _Float16* __restrict__ fWh,
                                                       float* __restrict__ h1) {
    const int tid  = blockIdx.x * blockDim.x + threadIdx.x;
    const int wid  = tid >> 5;
    const int lane = threadIdx.x & 31;

    const int ct = wid & 31;          // 32 c-tiles of 16
    const int bi = wid >> 5;          // 0 .. NB*NA-1
    const int i  = bi & (NA - 1);
    const int b  = bi >> 7;
    const int c0 = ct << 4;

    const int n    = lane & 15;       // row-in-tile (A) / column (B,C)
    const int half = lane >> 4;       // lane half selects K sub-block
    const int kb   = half * 8;

    // B tile: B[k, c] = filter_W[c0+n, k]; lane holds two 8-f16 K runs (kb.., 16+kb..)
    v16h bmat;
#pragma unroll
    for (int s = 0; s < 16; ++s)
        bmat[s] = fWh[(c0 + n) * 32 + (s < 8 ? kb + s : 16 + kb + (s - 8))];
    const float bias = filter_b[c0 + n];

    float acc = 0.0f;

    // incrementing base pointers; inner loads use small immediate offsets
    const float* erow = e + ((size_t)(b * NA + i) * NA + n) * EDGE_F + kb;
    const float* em   = edge_mask + (size_t)(b * NA + i) * NA + half * 8;
    const float* hcol = h + (size_t)(b * NA + half * 8) * CHANNEL + (c0 + n);

    // in-range offset for the two possibly-padded A elements (k = 30,31 when
    // half==1): read a clamped valid offset and cndmask to zero, no branches
    const int o67 = half ? 0 : 22;

#pragma unroll 2
    for (int j0 = 0; j0 < NA; j0 += 16) {
        __builtin_prefetch(erow + 16 * EDGE_F, 0, 1);   // next j-tile of e

        // ---- A tile: rows j0..j0+15 of e (K padded 30 -> 32) ----
        v16h amat;
#pragma unroll
        for (int t = 0; t < 8; ++t)
            amat[t] = (_Float16)erow[t];                // k = kb+t      (<=15)
#pragma unroll
        for (int t = 0; t < 6; ++t)
            amat[8 + t] = (_Float16)erow[16 + t];       // k = 16+kb+t   (<=29)
        float v6 = erow[o67];                           // k=22 (half0) / dummy
        float v7 = erow[o67 + 1];                       // k=23 (half0) / dummy
        amat[14] = half ? (_Float16)0.0f : (_Float16)v6;
        amat[15] = half ? (_Float16)0.0f : (_Float16)v7;

        v8f d = {};
        d = __builtin_amdgcn_wmma_f32_16x16x32_f16(false, amat, false, bmat,
                                                   (short)0, d, false, false);

        // ---- epilogue: selu + edge_mask, FMA against h, accumulate over j ----
        // C layout: VGPR r holds row j = j0 + half*8 + r, column c = c0 + n
#pragma unroll
        for (int r = 0; r < 8; ++r) {
            float ef = selu_f(d[r] + bias);
            float hm = hcol[(size_t)r * CHANNEL] * em[r];
            acc = fmaf(ef, hm, acc);
        }

        erow += 16 * EDGE_F;
        hcol += 16 * CHANNEL;
        em   += 16;
    }

    // combine the two lane-halves (same channel c0+n, disjoint j ranges)
    acc += __shfl_xor(acc, 16, 32);
    if (half == 0)
        h1[(size_t)(b * NA + i) * CHANNEL + c0 + n] = acc;
}

// ---------------- kernel 2: node MLP (2048 x 512 x 1024 GEMM) + epilogue ----
// out[m,c] = selu( concat(h,h1)[m,:] @ w_W[c,:] + w_b[c] ) * node_mask[m] + h[m,c]
__device__ __forceinline__ v8f mlp_k_block(const float* __restrict__ zrow,
                                           const _Float16* __restrict__ wrow,
                                           int half, v8f acc) {
    // A: two contiguous 8-float runs (32B-aligned: offsets are multiples of 8)
    const float4 q0 = ((const float4*)(zrow + half * 8))[0];
    const float4 q1 = ((const float4*)(zrow + half * 8))[1];
    const float4 q2 = ((const float4*)(zrow + 16 + half * 8))[0];
    const float4 q3 = ((const float4*)(zrow + 16 + half * 8))[1];

    v16h amat;
    amat[0]  = (_Float16)q0.x; amat[1]  = (_Float16)q0.y;
    amat[2]  = (_Float16)q0.z; amat[3]  = (_Float16)q0.w;
    amat[4]  = (_Float16)q1.x; amat[5]  = (_Float16)q1.y;
    amat[6]  = (_Float16)q1.z; amat[7]  = (_Float16)q1.w;
    amat[8]  = (_Float16)q2.x; amat[9]  = (_Float16)q2.y;
    amat[10] = (_Float16)q2.z; amat[11] = (_Float16)q2.w;
    amat[12] = (_Float16)q3.x; amat[13] = (_Float16)q3.y;
    amat[14] = (_Float16)q3.z; amat[15] = (_Float16)q3.w;

    v16h bmat;
#pragma unroll
    for (int s = 0; s < 16; ++s)
        bmat[s] = wrow[s < 8 ? half * 8 + s : 16 + half * 8 + (s - 8)];

    return __builtin_amdgcn_wmma_f32_16x16x32_f16(false, amat, false, bmat,
                                                  (short)0, acc, false, false);
}

__global__ __launch_bounds__(256) void node_mlp_kernel(const float* __restrict__ h,
                                                       const float* __restrict__ h1,
                                                       const float* __restrict__ node_mask,
                                                       const float* __restrict__ w_b,
                                                       const _Float16* __restrict__ wWh,
                                                       float* __restrict__ out) {
    const int tid  = blockIdx.x * blockDim.x + threadIdx.x;
    const int wid  = tid >> 5;
    const int lane = threadIdx.x & 31;

    const int nt = wid & 31;          // 32 N-tiles
    const int mt = wid >> 5;          // 128 M-tiles
    const int m0 = mt << 4;
    const int c0 = nt << 4;

    const int n    = lane & 15;
    const int half = lane >> 4;

    const float*    zrow_h  = h   + (size_t)(m0 + n) * CHANNEL;
    const float*    zrow_h1 = h1  + (size_t)(m0 + n) * CHANNEL;
    const _Float16* wrow    = wWh + (size_t)(c0 + n) * (2 * CHANNEL);

    v8f acc = {};

    // K split at the h|h1 seam (512 is a multiple of the 32-wide K tile),
    // so no per-iteration pointer selects.
#pragma unroll 4
    for (int k0 = 0; k0 < CHANNEL; k0 += 32)
        acc = mlp_k_block(zrow_h + k0, wrow + k0, half, acc);
#pragma unroll 4
    for (int k0 = 0; k0 < CHANNEL; k0 += 32)
        acc = mlp_k_block(zrow_h1 + k0, wrow + CHANNEL + k0, half, acc);

    // epilogue: bias + SELU + node_mask + residual
    const float bias = w_b[c0 + n];
    const int rowb = m0 + half * 8;
    const float* hres = h   + (size_t)rowb * CHANNEL + (c0 + n);
    float*       op   = out + (size_t)rowb * CHANNEL + (c0 + n);
#pragma unroll
    for (int r = 0; r < 8; ++r) {
        float nm = node_mask[rowb + r];
        op[(size_t)r * CHANNEL] =
            selu_f(acc[r] + bias) * nm + hres[(size_t)r * CHANNEL];
    }
}

// ---------------- launcher ----------------

extern "C" void kernel_launch(void* const* d_in, const int* in_sizes, int n_in,
                              void* d_out, int out_size, void* d_ws, size_t ws_size,
                              hipStream_t stream) {
    const float* h         = (const float*)d_in[0];
    const float* e         = (const float*)d_in[1];
    const float* node_mask = (const float*)d_in[2];
    const float* edge_mask = (const float*)d_in[3];
    const float* filter_W  = (const float*)d_in[4];
    const float* filter_b  = (const float*)d_in[5];
    const float* w_W       = (const float*)d_in[6];
    const float* w_b       = (const float*)d_in[7];
    float* out = (float*)d_out;

    // workspace layout
    char* ws = (char*)d_ws;
    _Float16* fWh = (_Float16*)(ws);                       // 512*32*2  = 32 KB
    _Float16* wWh = (_Float16*)(ws + 32768);               // 512*1024*2 = 1 MB
    float*    h1  = (float*)(ws + 32768 + 1048576);        // 16*128*512*4 = 4 MB

    // prep: weight conversion to f16
    cvt_filter_kernel<<<(CHANNEL * 32 + 255) / 256, 256, 0, stream>>>(filter_W, fWh);
    cvt_w_kernel<<<(CHANNEL * 2 * CHANNEL + 255) / 256, 256, 0, stream>>>(w_W, wWh);

    // fused edge filter + aggregation: NB*NA*(512/16) waves, 8 waves/block
    {
        int waves = NB * NA * (CHANNEL / 16);              // 65536
        edge_agg_kernel<<<waves * 32 / 256, 256, 0, stream>>>(e, h, edge_mask,
                                                              filter_b, fWh, h1);
    }

    // node MLP GEMM + epilogue: (2048/16)*(512/16) = 4096 waves
    {
        int waves = (NB * NA / 16) * (CHANNEL / 16);       // 4096
        node_mlp_kernel<<<waves * 32 / 256, 256, 0, stream>>>(h, h1, node_mask,
                                                              w_b, wWh, out);
    }
}
